// Seq2SeqLSTM_57526791962949
// MI455X (gfx1250) — matrix operand order repaired and checked by reading, not compiled, not run on hardware
//
#include <hip/hip_runtime.h>
#include <stdint.h>

// ---------------- types for WMMA fragments ----------------
typedef __bf16 bf16x16 __attribute__((ext_vector_type(16)));
typedef float  f32x8   __attribute__((ext_vector_type(8)));

union FragBF { bf16x16 v; uint4 q[2]; };  // 32 bytes = two b128 loads
union AccF   { f32x8 v; float f[8]; };

__device__ __forceinline__ uint4 ld128(const unsigned short* p) {
  return *reinterpret_cast<const uint4*>(p);
}

__device__ __forceinline__ unsigned short f2bf(float f) {
  union { float f; unsigned u; } v; v.f = f;
  unsigned u = v.u;
  u += 0x7FFFu + ((u >> 16) & 1u);   // round to nearest even
  return (unsigned short)(u >> 16);
}

__device__ __forceinline__ float sigmoidf_(float x) {
  return 1.0f / (1.0f + __expf(-x));
}

// ---------------- prep kernels ----------------

// in: [K][N] f32 row-major.  out: bf16, out[n*ostride + koff + k] = in[k][n]
__global__ void transpose_bf16(const float* __restrict__ in, unsigned short* __restrict__ out,
                               int K, int N, int ostride, int koff) {
  __shared__ float tile[32][33];
  const int nb = blockIdx.x * 32, kb = blockIdx.y * 32;
  const int tx = threadIdx.x, ty = threadIdx.y;   // 32 x 8
#pragma unroll
  for (int j = 0; j < 4; ++j) {
    int k = kb + ty + j * 8;
    tile[ty + j * 8][tx] = in[(size_t)k * N + nb + tx];
  }
  __syncthreads();
#pragma unroll
  for (int j = 0; j < 4; ++j) {
    int n = nb + ty + j * 8;
    out[(size_t)n * ostride + koff + kb + tx] = f2bf(tile[tx][ty + j * 8]);
  }
}

__global__ void cvt_f32_bf16(const float* __restrict__ in, unsigned short* __restrict__ out, int n) {
  int i = blockIdx.x * blockDim.x + threadIdx.x;
  if (i < n) out[i] = f2bf(in[i]);
}

// h0,c0: [3][1][1024].  hbuf: [3][2][64*1024] bf16 (parity 0 initialized). cbuf: [3][64*1024] f32
__global__ void init_states(const float* __restrict__ h0, const float* __restrict__ c0,
                            unsigned short* __restrict__ hbuf, float* __restrict__ cbuf) {
  int idx = blockIdx.x * 256 + threadIdx.x;      // 0 .. 3*64*1024-1
  if (idx >= 3 * 64 * 1024) return;
  int e   = idx >> 16;
  int rem = idx & 65535;
  int k   = rem & 1023;
  hbuf[(size_t)e * 2 * 65536 + rem] = f2bf(h0[e * 1024 + k]);
  cbuf[idx] = c0[e * 1024 + k];
}

// ---------------- fused LSTM layer-step kernel ----------------
// z = [x_in ; h_prev] @ Wcat + b, then LSTM cell with peephole Wc*c added to
// all 4 gates (reference quirk), updating c and h.
// Grid: 64 blocks (16 h-columns each). Block: 128 threads = 4 waves.
// Wave m_tile owns rows m_tile*16..+15 and keeps FOUR independent WMMA
// accumulator chains (one per gate) -> no hazard NOPs; register ping-pong
// prefetch overlaps next iteration's loads with current WMMAs.
template <int DIN>
__global__ __launch_bounds__(128) void lstm_layer_kernel(
    const unsigned short* __restrict__ xsrc, int xstride,
    const unsigned short* __restrict__ Wt,              // [4096][DIN+1024] bf16, N-major
    const unsigned short* __restrict__ hin,             // [64][1024] bf16
    const float* __restrict__ bias,                     // [4096] gate-major
    const float* __restrict__ Wc,                       // [4][1024]
    float* __restrict__ cbuf,                           // [64][1024] f32 (in/out)
    unsigned short* __restrict__ hout,                  // [64][1024] bf16
    unsigned short* __restrict__ extra)                 // optional extra copy of h
{
  constexpr int KT    = DIN + 1024;
  constexpr int NITER = KT / 32;
  __shared__ float zs[4][64][16];                       // 16 KB

  const int tid    = threadIdx.x;
  const int lane   = tid & 31;
  const int m_tile = tid >> 5;        // 0..3
  const int hl     = lane >> 4;       // lane half (WMMA K split)
  const int l15    = lane & 15;
  const int nb     = blockIdx.x << 4; // h-column base in [0,1024)
  const int row    = (m_tile << 4) + l15;

  // A bases (per-lane): x part and h part, chunk offset hl*8 baked in
  const unsigned short* ax = xsrc + (size_t)row * xstride + (hl << 3);
  const unsigned short* ah = hin  + (size_t)row * 1024    + (hl << 3);
  // B bases per gate: column-major rows of length KT, K offset hl*8 baked in
  const unsigned short* bp0 = Wt + (size_t)(nb + l15)        * KT + (hl << 3);
  const unsigned short* bp1 = Wt + (size_t)(1024 + nb + l15) * KT + (hl << 3);
  const unsigned short* bp2 = Wt + (size_t)(2048 + nb + l15) * KT + (hl << 3);
  const unsigned short* bp3 = Wt + (size_t)(3072 + nb + l15) * KT + (hl << 3);

  AccF ac0, ac1, ac2, ac3;
#pragma unroll
  for (int r = 0; r < 8; ++r) { ac0.f[r] = 0.f; ac1.f[r] = 0.f; ac2.f[r] = 0.f; ac3.f[r] = 0.f; }

  FragBF aC, b0C, b1C, b2C, b3C;
  // prologue: iteration 0 (always in the x part; DIN >= 32)
  aC.q[0]  = ld128(ax);       aC.q[1]  = ld128(ax + 16);
  b0C.q[0] = ld128(bp0);      b0C.q[1] = ld128(bp0 + 16);
  b1C.q[0] = ld128(bp1);      b1C.q[1] = ld128(bp1 + 16);
  b2C.q[0] = ld128(bp2);      b2C.q[1] = ld128(bp2 + 16);
  b3C.q[0] = ld128(bp3);      b3C.q[1] = ld128(bp3 + 16);

#pragma unroll
  for (int kt = 0; kt < NITER; ++kt) {
    FragBF aN, b0N, b1N, b2N, b3N;
    const bool more = (kt + 1) < NITER;
    if (more) {
      const int kb = (kt + 1) * 32;   // compile-time after unroll
      const unsigned short* ab = (kb < DIN) ? (ax + kb) : (ah + (kb - DIN));
      aN.q[0]  = ld128(ab);            aN.q[1]  = ld128(ab + 16);
      b0N.q[0] = ld128(bp0 + kb);      b0N.q[1] = ld128(bp0 + kb + 16);
      b1N.q[0] = ld128(bp1 + kb);      b1N.q[1] = ld128(bp1 + kb + 16);
      b2N.q[0] = ld128(bp2 + kb);      b2N.q[1] = ld128(bp2 + kb + 16);
      b3N.q[0] = ld128(bp3 + kb);      b3N.q[1] = ld128(bp3 + kb + 16);
    }
    // four independent accumulator chains: back-to-back WMMAs, no NOPs needed
    ac0.v = __builtin_amdgcn_wmma_f32_16x16x32_bf16(false, aC.v, false, b0C.v, (short)0, ac0.v, false, false);
    ac1.v = __builtin_amdgcn_wmma_f32_16x16x32_bf16(false, aC.v, false, b1C.v, (short)0, ac1.v, false, false);
    ac2.v = __builtin_amdgcn_wmma_f32_16x16x32_bf16(false, aC.v, false, b2C.v, (short)0, ac2.v, false, false);
    ac3.v = __builtin_amdgcn_wmma_f32_16x16x32_bf16(false, aC.v, false, b3C.v, (short)0, ac3.v, false, false);
    if (more) { aC = aN; b0C = b0N; b1C = b1N; b2C = b2N; b3C = b3N; }
  }

  // C/D layout: element r of lane -> M = m_tile*16 + hl*8 + r, N = l15
  {
    const int m0 = (m_tile << 4) + (hl << 3);
#pragma unroll
    for (int r = 0; r < 8; ++r) {
      zs[0][m0 + r][l15] = ac0.f[r];
      zs[1][m0 + r][l15] = ac1.f[r];
      zs[2][m0 + r][l15] = ac2.f[r];
      zs[3][m0 + r][l15] = ac3.f[r];
    }
  }
  __syncthreads();

  // ---- cell phase: 64 rows x 16 columns, 8 elements per thread ----
  for (int idx = tid; idx < 1024; idx += 128) {
    const int bb   = idx >> 4;
    const int nn   = idx & 15;
    const int hcol = nb + nn;
    const float cprev = cbuf[bb * 1024 + hcol];
    const float zf = zs[0][bb][nn] + bias[hcol]        + Wc[hcol]        * cprev;
    const float zi = zs[1][bb][nn] + bias[1024 + hcol] + Wc[1024 + hcol] * cprev;
    const float zg = zs[2][bb][nn] + bias[2048 + hcol] + Wc[2048 + hcol] * cprev;
    const float zo = zs[3][bb][nn] + bias[3072 + hcol] + Wc[3072 + hcol] * cprev;
    const float fg = sigmoidf_(zf);
    const float ig = sigmoidf_(zi);
    const float gg = tanhf(zg);
    const float og = sigmoidf_(zo);
    const float cn = fg * cprev + ig * gg;
    const float hn = og * tanhf(cn);
    cbuf[bb * 1024 + hcol] = cn;
    const unsigned short hb = f2bf(hn);
    hout[bb * 1024 + hcol] = hb;
    if (extra) extra[bb * 1024 + hcol] = hb;
  }
}

// ---------------- final linear + sigmoid ----------------
// out[r][d] = sigmoid(sum_k A[r][k]*Bt[d][k] + fb[d]); A=[4096][1024], Bt=[256][1024]
// Two split-K accumulator chains interleave independent WMMAs.
__global__ __launch_bounds__(128) void final_gemm_kernel(
    const unsigned short* __restrict__ A, const unsigned short* __restrict__ Bt,
    const float* __restrict__ fb, float* __restrict__ out)
{
  const int lane = threadIdx.x & 31;
  const int wave = threadIdx.x >> 5;   // 0..3 -> m_tile
  const int hl   = lane >> 4;
  const int l15  = lane & 15;
  const int mbase = blockIdx.y * 64 + wave * 16;
  const int ncol  = blockIdx.x * 16 + l15;
  const unsigned short* arow = A  + (size_t)(mbase + l15) * 1024 + (hl << 3);
  const unsigned short* brow = Bt + (size_t)ncol * 1024          + (hl << 3);

  AccF acA, acB;
#pragma unroll
  for (int r = 0; r < 8; ++r) { acA.f[r] = 0.f; acB.f[r] = 0.f; }

#pragma unroll
  for (int kt = 0; kt < 16; ++kt) {
    const int kb = kt * 32;
    FragBF a0, b0, a1, b1;
    a0.q[0] = ld128(arow + kb);        a0.q[1] = ld128(arow + kb + 16);
    b0.q[0] = ld128(brow + kb);        b0.q[1] = ld128(brow + kb + 16);
    a1.q[0] = ld128(arow + kb + 512);  a1.q[1] = ld128(arow + kb + 528);
    b1.q[0] = ld128(brow + kb + 512);  b1.q[1] = ld128(brow + kb + 528);
    acA.v = __builtin_amdgcn_wmma_f32_16x16x32_bf16(false, a0.v, false, b0.v, (short)0, acA.v, false, false);
    acB.v = __builtin_amdgcn_wmma_f32_16x16x32_bf16(false, a1.v, false, b1.v, (short)0, acB.v, false, false);
  }
  const float bv = fb[ncol];
#pragma unroll
  for (int r = 0; r < 8; ++r) {
    const int m = mbase + (hl << 3) + r;
    out[(size_t)m * 256 + ncol] = sigmoidf_(acA.f[r] + acB.f[r] + bv);
  }
}

// ---------------- host orchestration ----------------
extern "C" void kernel_launch(void* const* d_in, const int* in_sizes, int n_in,
                              void* d_out, int out_size, void* d_ws, size_t ws_size,
                              hipStream_t stream) {
  (void)in_sizes; (void)n_in; (void)out_size; (void)ws_size;
  const float* x       = (const float*)d_in[0];   // [64][64][256]
  const float* enc_Wx0 = (const float*)d_in[1];   // [256][4096]
  const float* enc_Wh0 = (const float*)d_in[2];   // [1024][4096]
  const float* enc_Wc0 = (const float*)d_in[3];   // [4][1024]
  const float* enc_b0  = (const float*)d_in[4];   // [4096]
  const float* enc_Wx  = (const float*)d_in[5];   // [2][1024][4096]
  const float* enc_Wh  = (const float*)d_in[6];   // [2][1024][4096]
  const float* enc_Wc  = (const float*)d_in[7];   // [2][4][1024]
  const float* enc_b   = (const float*)d_in[8];   // [2][4096]
  const float* dec_Wx  = (const float*)d_in[9];   // [3][1024][4096]
  const float* dec_Wh  = (const float*)d_in[10];  // [3][1024][4096]
  const float* dec_Wc  = (const float*)d_in[11];  // [3][4][1024]
  const float* dec_b   = (const float*)d_in[12];  // [3][4096]
  const float* fin_W   = (const float*)d_in[13];  // [1024][256]
  const float* fin_b   = (const float*)d_in[14];  // [256]
  const float* h0      = (const float*)d_in[15];  // [3][1][1024]
  const float* c0      = (const float*)d_in[16];  // [3][1][1024]
  // d_in[17] = pred_len (device scalar); fixed to 64 per setup_inputs.

  char* ws = (char*)d_ws;
  size_t off = 0;
  unsigned short* W[6];
  W[0] = (unsigned short*)(ws + off); off += (size_t)4096 * 1280 * 2;
  for (int i = 1; i < 6; ++i) { W[i] = (unsigned short*)(ws + off); off += (size_t)4096 * 2048 * 2; }
  unsigned short* finW = (unsigned short*)(ws + off); off += (size_t)256 * 1024 * 2;
  unsigned short* xbf  = (unsigned short*)(ws + off); off += (size_t)64 * 64 * 256 * 2;
  unsigned short* hbuf = (unsigned short*)(ws + off); off += (size_t)3 * 2 * 65536 * 2;
  float*          cbuf = (float*)(ws + off);          off += (size_t)3 * 65536 * 4;
  unsigned short* tops = (unsigned short*)(ws + off); off += (size_t)64 * 65536 * 2;

  auto Hb = [&](int e, int p) { return hbuf + ((size_t)e * 2 + p) * 65536; };
  auto Cb = [&](int e) { return cbuf + (size_t)e * 65536; };

  const dim3 tb(32, 8);
  // Weight prep: transpose to N-major bf16, Wx||Wh concatenated along K.
  transpose_bf16<<<dim3(128, 8),  tb, 0, stream>>>(enc_Wx0, W[0], 256, 4096, 1280, 0);
  transpose_bf16<<<dim3(128, 32), tb, 0, stream>>>(enc_Wh0, W[0], 1024, 4096, 1280, 256);
  for (int e = 0; e < 2; ++e) {
    transpose_bf16<<<dim3(128, 32), tb, 0, stream>>>(enc_Wx + (size_t)e * 1024 * 4096, W[1 + e], 1024, 4096, 2048, 0);
    transpose_bf16<<<dim3(128, 32), tb, 0, stream>>>(enc_Wh + (size_t)e * 1024 * 4096, W[1 + e], 1024, 4096, 2048, 1024);
  }
  for (int e = 0; e < 3; ++e) {
    transpose_bf16<<<dim3(128, 32), tb, 0, stream>>>(dec_Wx + (size_t)e * 1024 * 4096, W[3 + e], 1024, 4096, 2048, 0);
    transpose_bf16<<<dim3(128, 32), tb, 0, stream>>>(dec_Wh + (size_t)e * 1024 * 4096, W[3 + e], 1024, 4096, 2048, 1024);
  }
  transpose_bf16<<<dim3(8, 32), tb, 0, stream>>>(fin_W, finW, 1024, 256, 1024, 0);

  cvt_f32_bf16<<<dim3(4096), dim3(256), 0, stream>>>(x, xbf, 64 * 64 * 256);
  init_states<<<dim3(768), dim3(256), 0, stream>>>(h0, c0, hbuf, cbuf);

  const dim3 lg(64), lbk(128);
  // ---- encoder: 64 steps x 3 layers, state parity ping-pong ----
  for (int t = 0; t < 64; ++t) {
    const int p = t & 1, q = p ^ 1;
    lstm_layer_kernel<256><<<lg, lbk, 0, stream>>>(xbf + (size_t)t * 256, 16384, W[0],
        Hb(0, p), enc_b0, enc_Wc0, Cb(0), Hb(0, q), (unsigned short*)nullptr);
    lstm_layer_kernel<1024><<<lg, lbk, 0, stream>>>(Hb(0, q), 1024, W[1],
        Hb(1, p), enc_b, enc_Wc, Cb(1), Hb(1, q), (unsigned short*)nullptr);
    lstm_layer_kernel<1024><<<lg, lbk, 0, stream>>>(Hb(1, q), 1024, W[2],
        Hb(2, p), enc_b + 4096, enc_Wc + 4096, Cb(2), Hb(2, q), (unsigned short*)nullptr);
  }
  // ---- decoder: reuses the same 3 state slots; input = previous top h ----
  for (int s = 0; s < 64; ++s) {
    const int p = s & 1, q = p ^ 1;
    lstm_layer_kernel<1024><<<lg, lbk, 0, stream>>>(Hb(2, p), 1024, W[3],
        Hb(0, p), dec_b, dec_Wc, Cb(0), Hb(0, q), (unsigned short*)nullptr);
    lstm_layer_kernel<1024><<<lg, lbk, 0, stream>>>(Hb(0, q), 1024, W[4],
        Hb(1, p), dec_b + 4096, dec_Wc + 4096, Cb(1), Hb(1, q), (unsigned short*)nullptr);
    lstm_layer_kernel<1024><<<lg, lbk, 0, stream>>>(Hb(1, q), 1024, W[5],
        Hb(2, p), dec_b + 8192, dec_Wc + 8192, Cb(2), Hb(2, q), tops + (size_t)s * 65536);
  }
  // ---- final linear + sigmoid into d_out ([1,64,64,256]) ----
  final_gemm_kernel<<<dim3(16, 64), dim3(128), 0, stream>>>(tops, finW, fin_b, (float*)d_out);
}
